// Encoder_41455024341316
// MI455X (gfx1250) — matrix operand order repaired and checked by reading, not compiled, not run on hardware
//
#include <hip/hip_runtime.h>
#include <hip/hip_bf16.h>

typedef __attribute__((ext_vector_type(16))) _Float16 v16h;
typedef __attribute__((ext_vector_type(8)))  _Float16 v8h;
typedef __attribute__((ext_vector_type(4)))  _Float16 v4h;
typedef __attribute__((ext_vector_type(2)))  _Float16 v2h;
typedef __attribute__((ext_vector_type(8)))  float    v8f;

#define WAVES 8
#define BLOCK 256

__device__ __forceinline__ void lds_fence() {
  asm volatile("s_wait_dscnt 0" ::: "memory");
}

// 16 contiguous halfs (one pre-swizzled B fragment slot) -> two ds_load_b128
__device__ __forceinline__ v16h load_frag16(const _Float16* p) {
  v8h lo = *(const v8h*)p;
  v8h hi = *(const v8h*)(p + 8);
  return __builtin_shufflevector(lo, hi, 0,1,2,3,4,5,6,7,8,9,10,11,12,13,14,15);
}

// A-matrix 16x32 f16 fragment from row-major LDS tile (row stride 64 halfs).
// lane<16 -> M=lane, K={0..15}; lane>=16 -> M=lane-16, K={16..31}
// (K here is the per-layer *permuted* K index; B fragments use the same permutation.)
__device__ __forceinline__ v16h load_A_frag(const _Float16* s0, int koff, int lane) {
  int m   = lane & 15;
  int klo = (lane >> 4) << 4;                    // 0 or 16
  const _Float16* p = s0 + m * 64 + koff + klo;  // 16B aligned
  v8h lo = *(const v8h*)p;          // K = klo..klo+7
  v8h hi = *(const v8h*)(p + 8);    // K = klo+8..klo+15
  return __builtin_shufflevector(lo, hi, 0,1,2,3,4,5,6,7,8,9,10,11,12,13,14,15);
}

__device__ __forceinline__ v8f wmma_f16(v16h a, v16h b, v8f c) {
  return __builtin_amdgcn_wmma_f32_16x16x32_f16(false, a, false, b, (short)0, c, false, false);
}

template<bool EDGE>
__global__ __launch_bounds__(BLOCK) void enc_kernel(
    const float* __restrict__ pos, const long long* __restrict__ ei,
    const float* __restrict__ bounds,
    const float* __restrict__ W1, const float* __restrict__ b1,
    const float* __restrict__ W2, const float* __restrict__ b2,
    const float* __restrict__ W3, const float* __restrict__ b3,
    const float* __restrict__ g,  const float* __restrict__ bt,
    float* __restrict__ out, int nrows, int k1)
{
  // Weights pre-swizzled into B-fragment layout: frag slot = [(nc*KC+kc)*32 + lane]*16 + h
  // fragment-space k: kk = kc*32 + ((lane>>4)<<4) + h ; n = nc*16 + (lane&15)
  // W2 uses permuted K (h1 stored at k'=2n+c):  orig_k = (kk>>1) + 16*(kk&1)
  // W3 uses permuted K (h2 stored at k'=4n+t):  orig_k = (kk>>2) + 16*(kk&3)
  __shared__ _Float16 Ws1[2 * 512];        // 32(Kpad) x 32, identity K
  __shared__ _Float16 Ws2[4 * 512];        // 32 x 64, permuted K
  __shared__ _Float16 Ws3[16 * 512];       // 64 x 128, permuted K (8 n-tiles x 2 k-chunks)
  __shared__ _Float16 sA[WAVES * 16 * 64]; // per-wave activation staging

  for (int i = threadIdx.x; i < 2 * 512; i += BLOCK) {
    int h = i & 15, l = (i >> 4) & 31, nc = i >> 9;
    int nn = nc * 16 + (l & 15);
    int kk = ((l >> 4) << 4) + h;
    Ws1[i] = (kk < k1) ? (_Float16)W1[kk * 32 + nn] : (_Float16)0.f;
  }
  for (int i = threadIdx.x; i < 4 * 512; i += BLOCK) {
    int h = i & 15, l = (i >> 4) & 31, nc = i >> 9;
    int nn = nc * 16 + (l & 15);
    int kk = ((l >> 4) << 4) + h;
    int ok = (kk >> 1) + ((kk & 1) << 4);          // un-permute
    Ws2[i] = (_Float16)W2[ok * 64 + nn];
  }
  for (int i = threadIdx.x; i < 16 * 512; i += BLOCK) {
    int h = i & 15, l = (i >> 4) & 31, f = i >> 9;
    int kc = f & 1, nc = f >> 1;
    int nn = nc * 16 + (l & 15);
    int kk = kc * 32 + ((l >> 4) << 4) + h;
    int ok = (kk >> 2) + ((kk & 3) << 4);          // un-permute
    Ws3[i] = (_Float16)W3[ok * 128 + nn];
  }
  __syncthreads();

  const int lane = threadIdx.x & 31;
  const int wid  = threadIdx.x >> 5;
  const int n    = lane & 15;
  const int hi   = lane >> 4;
  _Float16* s0 = sA + wid * (16 * 64);

  // Loop-invariant B fragments for layers 1 & 2 held in registers (6 x 8 VGPRs)
  v16h B1r[2], B2r[4];
#pragma unroll
  for (int t = 0; t < 2; ++t) B1r[t] = load_frag16(Ws1 + (t * 32 + lane) * 16);
#pragma unroll
  for (int t = 0; t < 4; ++t) B2r[t] = load_frag16(Ws2 + (t * 32 + lane) * 16);

  // Per-lane bias / gamma / beta registers (loaded once from global)
  float b1r0 = b1[n], b1r1 = b1[16 + n];
  float b2r[4], b3r[8], gr[8], btr[8];
#pragma unroll
  for (int t = 0; t < 4; ++t) b2r[t] = b2[t * 16 + n];
#pragma unroll
  for (int t = 0; t < 8; ++t) { b3r[t] = b3[t * 16 + n]; gr[t] = g[t * 16 + n]; btr[t] = bt[t * 16 + n]; }

  const v8h zh = {};
  const v8f z  = {};

  const int ntiles = (nrows + 15) >> 4;
  for (int tile = blockIdx.x * WAVES + wid; tile < ntiles; tile += gridDim.x * WAVES) {
    // zero staging cols 16..31 (one b128 per lane); cols 0..15 fully written below
    *(v8h*)(s0 + (lane >> 1) * 64 + 16 + (lane & 1) * 8) = zh;

    if (lane < 16) {
      int r = tile * 16 + lane;
      if (r >= nrows) r = nrows - 1;
      float f[16];
#pragma unroll
      for (int j = 0; j < 16; ++j) f[j] = 0.f;
      if (EDGE) {
        long long si = ei[r], di = ei[(size_t)nrows + r];
        float4 psv = *(const float4*)(pos + (size_t)si * 12 + 8);  // {p8,p9,lastx,lasty}
        float4 pdv = *(const float4*)(pos + (size_t)di * 12 + 8);
        float dpx = psv.z - pdv.z, dpy = psv.w - pdv.w;
        float dist = sqrtf(dpx * dpx + dpy * dpy);
        float dsx = ((psv.z - psv.x) - (pdv.z - pdv.x)) * 1000.f;  // / VEL_STD
        float dsy = ((psv.w - psv.y) - (pdv.w - pdv.y)) * 1000.f;
        float ab  = sqrtf(dsx * dsx + dsy * dsy);
        float rel = (dsx * dpx + dsy * dpy) / (ab + 1e-6f) / (dist + 1e-6f);
        f[0] = dpx; f[1] = dpy; f[2] = dist; f[3] = dsx; f[4] = dsy; f[5] = ab; f[6] = rel;
      } else {
        float4 p0 = *(const float4*)(pos + (size_t)r * 12);
        float4 p1 = *(const float4*)(pos + (size_t)r * 12 + 4);
        float4 p2 = *(const float4*)(pos + (size_t)r * 12 + 8);
        float px[6] = {p0.x, p0.z, p1.x, p1.z, p2.x, p2.z};
        float py[6] = {p0.y, p0.w, p1.y, p1.w, p2.y, p2.w};
#pragma unroll
        for (int t = 0; t < 5; ++t) {
          f[2 * t]     = (px[t + 1] - px[t]) * 1000.f;
          f[2 * t + 1] = (py[t + 1] - py[t]) * 1000.f;
        }
        float lx = px[5], ly = py[5];
        f[10] = lx; f[11] = ly;
        f[12] = fminf(fmaxf(lx - bounds[0], -1.f), 1.f);
        f[13] = fminf(fmaxf(lx - bounds[1], -1.f), 1.f);
        f[14] = fminf(fmaxf(ly - bounds[2], -1.f), 1.f);
        f[15] = fminf(fmaxf(ly - bounds[3], -1.f), 1.f);
      }
      v8h r0, r1;
#pragma unroll
      for (int j = 0; j < 8; ++j) { r0[j] = (_Float16)f[j]; r1[j] = (_Float16)f[8 + j]; }
      _Float16* row = s0 + lane * 64;
      *(v8h*)row       = r0;
      *(v8h*)(row + 8) = r1;
    }
    lds_fence();

    // ---- Layer 1: [16x32] @ [32x32] (B in registers); h1 stored at k'=2n+c -> b32 stores
    v16h a = load_A_frag(s0, 0, lane);
    v8f d0 = wmma_f16(a, B1r[0], z);
    v8f d1 = wmma_f16(a, B1r[1], z);
#pragma unroll
    for (int v = 0; v < 8; ++v) {
      int m = v + hi * 8;
      v2h pk;
      pk[0] = (_Float16)fmaxf(d0[v] + b1r0, 0.f);
      pk[1] = (_Float16)fmaxf(d1[v] + b1r1, 0.f);
      *(v2h*)(s0 + m * 64 + 2 * n) = pk;
    }
    lds_fence();

    // ---- Layer 2: [16x32] @ [32x64] (B in registers); h2 stored at k'=4n+t -> b64 stores
    a = load_A_frag(s0, 0, lane);
    v8f e0 = wmma_f16(a, B2r[0], z);
    v8f e1 = wmma_f16(a, B2r[1], z);
    v8f e2 = wmma_f16(a, B2r[2], z);
    v8f e3 = wmma_f16(a, B2r[3], z);
#pragma unroll
    for (int v = 0; v < 8; ++v) {
      int m = v + hi * 8;
      v4h pk;
      pk[0] = (_Float16)fmaxf(e0[v] + b2r[0], 0.f);
      pk[1] = (_Float16)fmaxf(e1[v] + b2r[1], 0.f);
      pk[2] = (_Float16)fmaxf(e2[v] + b2r[2], 0.f);
      pk[3] = (_Float16)fmaxf(e3[v] + b2r[3], 0.f);
      *(v4h*)(s0 + m * 64 + 4 * n) = pk;
    }
    lds_fence();

    // ---- Layer 3: [16x64] @ [64x128] (B fragments from pre-swizzled LDS)
    v16h a0 = load_A_frag(s0, 0,  lane);
    v16h a1 = load_A_frag(s0, 32, lane);
    v8f acc[8];
#pragma unroll
    for (int t = 0; t < 8; ++t) {
      acc[t] = wmma_f16(a0, load_frag16(Ws3 + ((t * 2 + 0) * 32 + lane) * 16), z);
      acc[t] = wmma_f16(a1, load_frag16(Ws3 + ((t * 2 + 1) * 32 + lane) * 16), acc[t]);
#pragma unroll
      for (int v = 0; v < 8; ++v) acc[t][v] += b3r[t];
    }

    // ---- LayerNorm: row's 128 values live across a 16-lane half -> shfl_xor reduce
#pragma unroll
    for (int v = 0; v < 8; ++v) {
      float s = 0.f, s2 = 0.f;
#pragma unroll
      for (int t = 0; t < 8; ++t) { float x = acc[t][v]; s += x; s2 += x * x; }
#pragma unroll
      for (int msk = 1; msk < 16; msk <<= 1) {
        s  += __shfl_xor(s,  msk, 32);
        s2 += __shfl_xor(s2, msk, 32);
      }
      float mu  = s * (1.f / 128.f);
      float var = s2 * (1.f / 128.f) - mu * mu;
      float rs  = rsqrtf(var + 1e-5f);
#pragma unroll
      for (int t = 0; t < 8; ++t)
        acc[t][v] = (acc[t][v] - mu) * rs * gr[t] + btr[t];
    }

    // ---- Store (lanes 0-15 cover 64 contiguous bytes per row segment)
    float* outRow = out + (size_t)tile * 16 * 128;
    int rowsLeft = nrows - tile * 16;
    if (rowsLeft >= 16) {
#pragma unroll
      for (int t = 0; t < 8; ++t)
#pragma unroll
        for (int v = 0; v < 8; ++v)
          outRow[(size_t)(v + hi * 8) * 128 + t * 16 + n] = acc[t][v];
    } else {
#pragma unroll
      for (int t = 0; t < 8; ++t)
#pragma unroll
        for (int v = 0; v < 8; ++v) {
          int m = v + hi * 8;
          if (m < rowsLeft) outRow[(size_t)m * 128 + t * 16 + n] = acc[t][v];
        }
    }
  }
}

extern "C" void kernel_launch(void* const* d_in, const int* in_sizes, int n_in,
                              void* d_out, int out_size, void* d_ws, size_t ws_size,
                              hipStream_t stream) {
  const float*     pos    = (const float*)d_in[0];
  const long long* ei     = (const long long*)d_in[1];   // int64 edge_index
  const float*     bounds = (const float*)d_in[2];
  const float* W1  = (const float*)d_in[3];  const float* b1  = (const float*)d_in[4];
  const float* W2  = (const float*)d_in[5];  const float* b2  = (const float*)d_in[6];
  const float* W3  = (const float*)d_in[7];  const float* b3  = (const float*)d_in[8];
  const float* g   = (const float*)d_in[9];  const float* bt  = (const float*)d_in[10];
  const float* eW1 = (const float*)d_in[11]; const float* eb1 = (const float*)d_in[12];
  const float* eW2 = (const float*)d_in[13]; const float* eb2 = (const float*)d_in[14];
  const float* eW3 = (const float*)d_in[15]; const float* eb3 = (const float*)d_in[16];
  const float* eg  = (const float*)d_in[17]; const float* ebt = (const float*)d_in[18];

  const int N = in_sizes[0] / 12;  // (N, 6, 2)
  const int E = in_sizes[1] / 2;   // (2, E)
  float* out = (float*)d_out;

  int ntN  = (N + 15) / 16;
  int blkN = (ntN + WAVES - 1) / WAVES; if (blkN > 1024) blkN = 1024;
  enc_kernel<false><<<blkN, BLOCK, 0, stream>>>(
      pos, nullptr, bounds, W1, b1, W2, b2, W3, b3, g, bt, out, N, 16);

  int ntE  = (E + 15) / 16;
  int blkE = (ntE + WAVES - 1) / WAVES; if (blkE > 4096) blkE = 4096;
  enc_kernel<true><<<blkE, BLOCK, 0, stream>>>(
      pos, ei, bounds, eW1, eb1, eW2, eb2, eW3, eb3, eg, ebt,
      out + (size_t)N * 128, E, 7);
}
